// ESNReservoir_15668040695916
// MI455X (gfx1250) — compile-verified, matrix operand order checked by reading, not run on hardware
//
#include <hip/hip_runtime.h>

#define RESV   4096
#define NIN    8
#define NOUT   8
#define WARM   1023     // steps 0..1022 teacher-only; step 1023 uses x[1023] and emits out[0]
#define TOTAL  1279     // 1023 + 256
#define KCH    128      // K chunks of 32
#define KHALF  64       // chunks per wave (2-way K split)
#define TPB    256
#define NWG_TILE 64     // 64 blocks * 8 waves = 512 waves = 256 tiles * 2 k-halves
#define NWG    65       // + 1 readout block
#define SLOPE  0.01f    // jax.nn.leaky_relu default

typedef __attribute__((ext_vector_type(16))) __bf16 v16bf;
typedef __attribute__((ext_vector_type(8)))  float  v8f;
typedef __attribute__((ext_vector_type(4)))  unsigned int v4u;

union ABits { v4u u[2]; v16bf v; };

__device__ __forceinline__ unsigned short f2bf(float f) {
  union { float f; unsigned u; } c; c.f = f;
  unsigned u = c.u;
  u += 0x7fffu + ((u >> 16) & 1u);      // round-to-nearest-even
  return (unsigned short)(u >> 16);
}

// Software grid barrier (65 WGs of 256 threads are trivially co-resident).
__device__ __forceinline__ void grid_sync(unsigned* cnt, unsigned* gen) {
  __threadfence();
  __syncthreads();
  if (threadIdx.x == 0) {
    unsigned g = __hip_atomic_load(gen, __ATOMIC_RELAXED, __HIP_MEMORY_SCOPE_AGENT);
    unsigned a = __hip_atomic_fetch_add(cnt, 1u, __ATOMIC_ACQ_REL, __HIP_MEMORY_SCOPE_AGENT);
    if (a == NWG - 1) {
      __hip_atomic_store(cnt, 0u, __ATOMIC_RELAXED, __HIP_MEMORY_SCOPE_AGENT);
      __hip_atomic_fetch_add(gen, 1u, __ATOMIC_ACQ_REL, __HIP_MEMORY_SCOPE_AGENT);
    } else {
      while (__hip_atomic_load(gen, __ATOMIC_ACQUIRE, __HIP_MEMORY_SCOPE_AGENT) == g)
        __builtin_amdgcn_s_sleep(1);
    }
  }
  __syncthreads();
}

// Pack W (fp32 row-major) into bf16 in the exact wave32 A-operand layout of
// v_wmma_f32_16x16x32_bf16: block (tile T, kchunk c) is 512 bf16 = 1KB
// contiguous; within a block, lane l owns 16 contiguous bf16 (32B):
//   lanes 0-15 : row 16T+l, K = 32c+{0..7}  then 32c+{16..23}
//   lanes 16-31: row 16T+(l-16), K = 32c+{8..15} then 32c+{24..31}
__global__ void esn_pack_w(const float* __restrict__ W, unsigned short* __restrict__ Wp) {
  const int stride = gridDim.x * blockDim.x;
  for (int i = blockIdx.x * blockDim.x + threadIdx.x; i < RESV * RESV; i += stride) {
    int e    = i & 511;
    int blk  = i >> 9;
    int c    = blk & (KCH - 1);
    int T    = blk >> 7;
    int lane = e >> 4;
    int j    = e & 15;
    int row  = T * 16 + (lane & 15);
    int col  = c * 32 + (lane >> 4) * 8 + (j & 7) + ((j >> 3) << 4);
    Wp[i] = f2bf(W[(size_t)row * RESV + col]);
  }
}

__global__ void esn_init(unsigned short* hbf, float* hf, float* inp, unsigned* ctrl) {
  int i = blockIdx.x * blockDim.x + threadIdx.x;
  int stride = gridDim.x * blockDim.x;
  for (int k = i; k < 2 * RESV; k += stride) { hbf[k] = 0; hf[k] = 0.f; }
  if (i < NIN) inp[i] = 0.f;
  if (i < 32) ctrl[i] = 0u;
}

__global__ __launch_bounds__(TPB)
void esn_run(const float* __restrict__ x,        // [1024][8]
             const float* __restrict__ Win,      // [4096][8]
             const float* __restrict__ Wout_w,   // [8][4096]
             const float* __restrict__ Wout_b,   // [8]
             const unsigned short* __restrict__ Wp,
             unsigned short* __restrict__ hbf,   // [2][4096] bf16 (WMMA B feed)
             float* __restrict__ hf,             // [2][4096] f32  (readout feed)
             float* __restrict__ inp_buf,        // [8] fed-back input
             unsigned* __restrict__ ctrl,
             float* __restrict__ out)            // [256][8]
{
  __shared__ float lds_part[8][16];
  const int tid   = threadIdx.x;
  const int lane  = tid & 31;
  const int wv    = tid >> 5;           // wave in WG: 0..7
  const int laneL = lane & 15;
  const int hi    = lane >> 4;
  const bool tileBlock = (blockIdx.x < NWG_TILE);
  const int  T  = blockIdx.x * 4 + (wv >> 1);   // output tile (16 rows)
  const int  kh = wv & 1;                       // which K half

  unsigned mbit = (laneL == 0) ? 0xFFFFFFFFu : 0u;  // B has only column 0 live
  const v4u msk = {mbit, mbit, mbit, mbit};

  unsigned* cnt = ctrl;
  unsigned* gen = ctrl + 16;

  for (int t = 0; t < TOTAL; ++t) {
    const int cur = t & 1;
    const int nxt = cur ^ 1;

    // ---- phase 1: h_new = leaky_relu(Win@inp + W@h) via WMMA ----
    if (tileBlock) {
      const unsigned short* wp = Wp + (((size_t)T * KCH + kh * KHALF) << 9) + lane * 16;
      const unsigned short* hb = hbf + cur * RESV + kh * (KHALF * 32);
      v8f acc = {0.f, 0.f, 0.f, 0.f, 0.f, 0.f, 0.f, 0.f};
      #pragma unroll 4
      for (int c = 0; c < KHALF; ++c) {
        ABits A, B;
        A.u[0] = *(const v4u*)(wp);          // 2x global_load_b128, fully coalesced
        A.u[1] = *(const v4u*)(wp + 8);
        const unsigned short* hp = hb + c * 32 + hi * 8;
        B.u[0] = *(const v4u*)(hp)      & msk;  // h segment in column 0 (lanes 0/16)
        B.u[1] = *(const v4u*)(hp + 16) & msk;
        acc = __builtin_amdgcn_wmma_f32_16x16x32_bf16(false, A.v, false, B.v,
                                                      (short)0, acc, false, false);
        wp += 512;
      }
      // D column 0: lane 0 holds rows T*16+0..7 (acc[0..7]), lane 16 rows +8..15
      if (laneL == 0) {
        #pragma unroll
        for (int i = 0; i < 8; ++i) lds_part[wv][hi * 8 + i] = acc[i];
      }
    }
    __syncthreads();
    if (tileBlock && kh == 0 && laneL == 0) {
      float in_v[NIN];
      const float* src = (t <= WARM) ? (x + t * NIN) : inp_buf;
      #pragma unroll
      for (int j = 0; j < NIN; ++j) in_v[j] = src[j];
      #pragma unroll
      for (int i = 0; i < 8; ++i) {
        int r   = hi * 8 + i;
        int row = T * 16 + r;
        float y = lds_part[wv][r] + lds_part[wv + 1][r];   // combine K halves
        const float* wr = Win + row * NIN;
        #pragma unroll
        for (int j = 0; j < NIN; ++j) y = fmaf(wr[j], in_v[j], y);
        y = (y > 0.f) ? y : SLOPE * y;                     // leaky_relu(0.01)
        hf [nxt * RESV + row] = y;
        hbf[nxt * RESV + row] = f2bf(y);
      }
    }
    grid_sync(cnt, gen);

    // ---- phase 2 (AR steps only): out = Wout@h_new + b, feed back ----
    if (t >= WARM) {
      if (!tileBlock) {
        const int o = wv;                     // one wave per output channel
        const float* wr = Wout_w + o * RESV;
        const float* h  = hf + nxt * RESV;
        float s = 0.f;
        for (int k = lane; k < RESV; k += 32) s = fmaf(wr[k], h[k], s);
        #pragma unroll
        for (int d = 16; d > 0; d >>= 1) s += __shfl_down(s, d, 32);
        if (lane == 0) {
          float ov = s + Wout_b[o];
          out[(t - WARM) * NOUT + o] = ov;
          inp_buf[o] = ov;
        }
      }
      grid_sync(cnt, gen);
    }
  }
}

extern "C" void kernel_launch(void* const* d_in, const int* in_sizes, int n_in,
                              void* d_out, int out_size, void* d_ws, size_t ws_size,
                              hipStream_t stream) {
  const float* x      = (const float*)d_in[0];
  const float* Win    = (const float*)d_in[1];
  const float* W      = (const float*)d_in[2];
  const float* Wout_w = (const float*)d_in[3];
  const float* Wout_b = (const float*)d_in[4];
  float* out = (float*)d_out;

  char* ws = (char*)d_ws;
  size_t off = 0;
  unsigned short* Wp  = (unsigned short*)(ws + off); off += (size_t)RESV * RESV * 2; // 32 MB packed bf16 W
  unsigned short* hbf = (unsigned short*)(ws + off); off += (size_t)2 * RESV * 2;    // 16 KB
  float*          hf  = (float*)(ws + off);          off += (size_t)2 * RESV * 4;    // 32 KB
  float*          inp = (float*)(ws + off);          off += 256;
  unsigned*       ctl = (unsigned*)(ws + off);       off += 256;

  esn_pack_w<<<dim3(2048), dim3(256), 0, stream>>>(W, Wp);
  esn_init  <<<dim3(32),   dim3(256), 0, stream>>>(hbf, hf, inp, ctl);
  esn_run   <<<dim3(NWG),  dim3(TPB), 0, stream>>>(x, Win, Wout_w, Wout_b,
                                                   Wp, hbf, hf, inp, ctl, out);
}
